// DecentralizedCoordinator_53523882443182
// MI455X (gfx1250) — compile-verified, matrix operand order checked by reading, not run on hardware
//
#include <hip/hip_runtime.h>
#include <hip/hip_bf16.h>
#include <math.h>

// ---------------------------------------------------------------------------
// DecentralizedCoordinator: N=100000 nodes, E=800000 edges, H=C=128
//   1) logits = x @ w_lead + b_lead                      (GEMV, mem-bound)
//   2) leader election: packed-u64 atomicMax per dst     (scatter argmax)
//   3) scatter-mean of x over edges                      (atomic f32, L2-resident)
//   4) reports = gelu(mean @ w1 + b1) @ w2 + b2          (WMMA bf16->f32)
//   5) out[i] = reports[leader[i]]                       (gather)
// ---------------------------------------------------------------------------

typedef __attribute__((ext_vector_type(16))) __bf16 v16bf;
typedef __attribute__((ext_vector_type(8)))  __bf16 v8bf;
typedef __attribute__((ext_vector_type(8)))  float  v8f;

#define HDIM 128

// ---- ordered-float key: monotonic uint encoding of float -------------------
__device__ __forceinline__ unsigned int float_key(float f) {
    unsigned int b = __float_as_uint(f);
    return (b & 0x80000000u) ? ~b : (b | 0x80000000u);
}

// ---- kernel 0: init workspace ---------------------------------------------
__global__ void dc_init_kernel(float* __restrict__ sums, float* __restrict__ cnt,
                               unsigned long long* __restrict__ packed, int n) {
    int i = blockIdx.x * blockDim.x + threadIdx.x;
    if (i < n * HDIM) sums[i] = 0.0f;
    if (i < n) { cnt[i] = 0.0f; packed[i] = 0ull; }
}

// ---- kernel 1: logits = x @ w_lead + b  (one wave32 per row) ---------------
__global__ void dc_logits_kernel(const float* __restrict__ x,
                                 const float* __restrict__ wl,
                                 const float* __restrict__ bl,
                                 float* __restrict__ logits, int n) {
    int w    = (blockIdx.x * blockDim.x + threadIdx.x) >> 5;
    int lane = threadIdx.x & 31;
    if (w >= n) return;
    float4 xv = ((const float4*)(x + (size_t)w * HDIM))[lane];
    float4 wv = ((const float4*)wl)[lane];
    float s = xv.x * wv.x + xv.y * wv.y + xv.z * wv.z + xv.w * wv.w;
    #pragma unroll
    for (int o = 16; o > 0; o >>= 1) s += __shfl_down(s, o, 32);
    if (lane == 0) logits[w] = s + bl[0];
}

// ---- kernel 2: leader election over E original + N self-loop edges ---------
__global__ void dc_leader_kernel(const int* __restrict__ ei,
                                 const float* __restrict__ logits,
                                 unsigned long long* __restrict__ packed,
                                 int E, int n) {
    int i = blockIdx.x * blockDim.x + threadIdx.x;
    if (i >= E + n) return;
    int src, dst;
    if (i < E) { src = ei[i]; dst = ei[i + E]; }
    else       { src = i - E; dst = src; }
    unsigned long long p =
        ((unsigned long long)float_key(logits[src]) << 32) | (unsigned int)src;
    atomicMax(packed + dst, p);
}

// ---- kernel 3: scatter-add x[src] -> sums[dst], cnt[dst]++ (wave per edge) -
__global__ void dc_scatter_kernel(const float* __restrict__ x,
                                  const int* __restrict__ ei,
                                  float* __restrict__ sums,
                                  float* __restrict__ cnt, int E) {
    int e    = (blockIdx.x * blockDim.x + threadIdx.x) >> 5;
    int lane = threadIdx.x & 31;
    if (e >= E) return;
    int src = ei[e], dst = ei[e + E];
    float4 v = ((const float4*)(x + (size_t)src * HDIM))[lane];
    float* dp = sums + (size_t)dst * HDIM + lane * 4;
    atomicAdd(dp + 0, v.x);
    atomicAdd(dp + 1, v.y);
    atomicAdd(dp + 2, v.z);
    atomicAdd(dp + 3, v.w);
    if (lane == 0) atomicAdd(cnt + dst, 1.0f);
}

// ---- kernel 4: in-place mean ----------------------------------------------
__global__ void dc_mean_kernel(float* __restrict__ sums,
                               const float* __restrict__ cnt, int n) {
    int i = blockIdx.x * blockDim.x + threadIdx.x;
    if (i >= n * HDIM) return;
    float c = cnt[i >> 7];
    c = c < 1.0f ? 1.0f : c;
    sums[i] *= (1.0f / c);
}

// ---- kernel 5: fused MLP with WMMA bf16 -----------------------------------
// Block = 256 threads = 8 waves; 128 rows per block, each wave a 16-row
// M-tile. Weights are pre-packed into LDS in WMMA B-fragment layout
// lds[mat][kc][nt][lane][16] so each fragment is ONE contiguous 32-byte
// per-lane LDS read (-> 2x ds_load_b128). The packed-w1 region (exactly
// 16384 bf16 = 32 KB) is reused as the 128x128 row-major hidden tile.
//
// WMMA wave32 layouts (CDNA5 ISA):
//   A/B 16-bit frag, lane l: row/col = l&15; elem e: k = e<8 ? kb+e : 16+kb+e-8
//   where kb = (l>>4)*8.  C/D f32: VGPR r -> M = r + 8*(l>>4), N = l&15.
__device__ __forceinline__ int wmma_kmap(int e, int kBase) {
    return (e < 8) ? (kBase + e) : (16 + kBase + (e - 8));
}

__global__ void __launch_bounds__(256)
dc_mlp_wmma_kernel(const float* __restrict__ nm,
                   const float* __restrict__ w1, const float* __restrict__ b1,
                   const float* __restrict__ w2, const float* __restrict__ b2,
                   float* __restrict__ reports, int n) {
    // [matrix][kc][nt][lane][e] : 2*4*8*32*16 bf16 = 64 KB
    __shared__ __bf16 lds[2][4][8][32][16];

    const int tid = threadIdx.x;

    // ---- pack w1/w2 into fragment-ready LDS layout ----
    // 2048 lane-fragments total; 8 per thread, 16 elems each.
    for (int idx = tid; idx < 2 * 4 * 8 * 32; idx += 256) {
        int lane = idx & 31;
        int nt   = (idx >> 5) & 7;
        int kcc  = (idx >> 8) & 3;
        int mat  = idx >> 10;
        const float* w = mat ? w2 : w1;
        int kb  = (lane >> 4) << 3;
        int col = nt * 16 + (lane & 15);
        __bf16* dst = &lds[mat][kcc][nt][lane][0];
        #pragma unroll
        for (int e = 0; e < 16; ++e)
            dst[e] = (__bf16)w[(kcc * 32 + wmma_kmap(e, kb)) * HDIM + col];
    }
    __syncthreads();

    const int wave  = tid >> 5;
    const int lane  = tid & 31;
    const int mrow  = lane & 15;
    const int hi    = lane >> 4;      // 0 or 1
    const int kBase = hi << 3;        // 0 or 8
    const int m0    = blockIdx.x * 128 + wave * 16;

    // -------- GEMM1: acc = mean @ w1 + b1 ----------
    v8f acc[8];
    #pragma unroll
    for (int nt = 0; nt < 8; ++nt) {
        float bv = b1[nt * 16 + mrow];
        #pragma unroll
        for (int r = 0; r < 8; ++r) acc[nt][r] = bv;
    }

    const int  arow = m0 + mrow;
    const bool inb  = arow < n;
    const float* aptr = nm + (size_t)(inb ? arow : 0) * HDIM;

    #pragma unroll
    for (int kc = 0; kc < 4; ++kc) {
        // A fragment: two contiguous 8-float runs -> 4x global_load_b128
        float4 f0 = *(const float4*)(aptr + kc * 32 + kBase);
        float4 f1 = *(const float4*)(aptr + kc * 32 + kBase + 4);
        float4 f2 = *(const float4*)(aptr + kc * 32 + 16 + kBase);
        float4 f3 = *(const float4*)(aptr + kc * 32 + 16 + kBase + 4);
        float sel = inb ? 1.0f : 0.0f;   // uniform per lane, no divergence
        v16bf a;
        a[0]  = (__bf16)(f0.x * sel); a[1]  = (__bf16)(f0.y * sel);
        a[2]  = (__bf16)(f0.z * sel); a[3]  = (__bf16)(f0.w * sel);
        a[4]  = (__bf16)(f1.x * sel); a[5]  = (__bf16)(f1.y * sel);
        a[6]  = (__bf16)(f1.z * sel); a[7]  = (__bf16)(f1.w * sel);
        a[8]  = (__bf16)(f2.x * sel); a[9]  = (__bf16)(f2.y * sel);
        a[10] = (__bf16)(f2.z * sel); a[11] = (__bf16)(f2.w * sel);
        a[12] = (__bf16)(f3.x * sel); a[13] = (__bf16)(f3.y * sel);
        a[14] = (__bf16)(f3.z * sel); a[15] = (__bf16)(f3.w * sel);

        #pragma unroll
        for (int nt = 0; nt < 8; ++nt) {
            v16bf b = *(const v16bf*)&lds[0][kc][nt][lane][0];  // 2x ds_load_b128
            acc[nt] = __builtin_amdgcn_wmma_f32_16x16x32_bf16(
                false, a, false, b, (short)0, acc[nt], false, false);
        }
    }

    __syncthreads();  // everyone done reading packed w1

    // -------- GELU (exact) + stash hidden tile (row-major) in lds[0] -------
    __bf16* hrows = &lds[0][0][0][0][0];   // reused: [128 rows][128 cols]
    #pragma unroll
    for (int nt = 0; nt < 8; ++nt) {
        #pragma unroll
        for (int r = 0; r < 8; ++r) {
            float v = acc[nt][r];
            float g = 0.5f * v * (1.0f + erff(v * 0.70710678118654752f));
            int rowl = wave * 16 + r + 8 * hi;
            hrows[rowl * HDIM + nt * 16 + mrow] = (__bf16)g;
        }
    }
    __syncthreads();

    // -------- GEMM2: acc2 = h @ w2 + b2 ----------
    v8f acc2[8];
    #pragma unroll
    for (int nt = 0; nt < 8; ++nt) {
        float bv = b2[nt * 16 + mrow];
        #pragma unroll
        for (int r = 0; r < 8; ++r) acc2[nt][r] = bv;
    }

    const __bf16* hp = hrows + (wave * 16 + mrow) * HDIM;
    #pragma unroll
    for (int kc = 0; kc < 4; ++kc) {
        // A fragment from row-major h: two 16-byte LDS loads
        v8bf lo  = *(const v8bf*)(hp + kc * 32 + kBase);
        v8bf hiv = *(const v8bf*)(hp + kc * 32 + 16 + kBase);
        v16bf a = __builtin_shufflevector(lo, hiv,
                    0, 1, 2, 3, 4, 5, 6, 7, 8, 9, 10, 11, 12, 13, 14, 15);
        #pragma unroll
        for (int nt = 0; nt < 8; ++nt) {
            v16bf b = *(const v16bf*)&lds[1][kc][nt][lane][0];  // 2x ds_load_b128
            acc2[nt] = __builtin_amdgcn_wmma_f32_16x16x32_bf16(
                false, a, false, b, (short)0, acc2[nt], false, false);
        }
    }

    // -------- store reports ----------
    #pragma unroll
    for (int nt = 0; nt < 8; ++nt) {
        #pragma unroll
        for (int r = 0; r < 8; ++r) {
            int grow = m0 + r + 8 * hi;
            if (grow < n)
                reports[(size_t)grow * HDIM + nt * 16 + mrow] = acc2[nt][r];
        }
    }
}

// ---- kernel 6: out[i] = reports[leader[i]]  (wave per row, float4) ---------
__global__ void dc_gather_kernel(const float* __restrict__ reports,
                                 const unsigned long long* __restrict__ packed,
                                 float* __restrict__ out, int n) {
    int w    = (blockIdx.x * blockDim.x + threadIdx.x) >> 5;
    int lane = threadIdx.x & 31;
    if (w >= n) return;
    int leader = (int)(packed[w] & 0xFFFFFFFFull);
    const float4* rp = (const float4*)(reports + (size_t)leader * HDIM);
    float4* op = (float4*)(out + (size_t)w * HDIM);
    op[lane] = rp[lane];
}

// ---------------------------------------------------------------------------
extern "C" void kernel_launch(void* const* d_in, const int* in_sizes, int n_in,
                              void* d_out, int out_size, void* d_ws, size_t ws_size,
                              hipStream_t stream) {
    const float* x      = (const float*)d_in[0];
    const int*   ei     = (const int*)d_in[1];
    const float* w_lead = (const float*)d_in[2];
    const float* b_lead = (const float*)d_in[3];
    const float* w1     = (const float*)d_in[4];
    const float* b1     = (const float*)d_in[5];
    const float* w2     = (const float*)d_in[6];
    const float* b2     = (const float*)d_in[7];

    const int N = in_sizes[0] / HDIM;   // 100000
    const int E = in_sizes[1] / 2;      // 800000

    // workspace layout (all offsets 8B-aligned)
    float* sums    = (float*)d_ws;                  // N*128 f32
    float* reports = sums + (size_t)N * HDIM;       // N*128 f32
    float* logits  = reports + (size_t)N * HDIM;    // N f32
    float* cnt     = logits + N;                    // N f32
    unsigned long long* packed = (unsigned long long*)(cnt + N);  // N u64

    float* out = (float*)d_out;

    const int T = 256;

    // 0) init sums/cnt/packed
    dc_init_kernel<<<(N * HDIM + T - 1) / T, T, 0, stream>>>(sums, cnt, packed, N);
    // 1) logits (one wave per row -> N*32 threads)
    dc_logits_kernel<<<(N * 32 + T - 1) / T, T, 0, stream>>>(x, w_lead, b_lead, logits, N);
    // 2) leader election over E + N extended edges
    dc_leader_kernel<<<(E + N + T - 1) / T, T, 0, stream>>>(ei, logits, packed, E, N);
    // 3) scatter-add (one wave per edge)
    dc_scatter_kernel<<<(E * 32 + T - 1) / T, T, 0, stream>>>(x, ei, sums, cnt, E);
    // 4) mean in place
    dc_mean_kernel<<<(N * HDIM + T - 1) / T, T, 0, stream>>>(sums, cnt, N);
    // 5) fused WMMA MLP, 128 rows per block
    dc_mlp_wmma_kernel<<<(N + 127) / 128, T, 0, stream>>>(sums, w1, b1, w2, b2, reports, N);
    // 6) gather leader reports (one wave per row)
    dc_gather_kernel<<<(N * 32 + T - 1) / T, T, 0, stream>>>(reports, packed, out, N);
}
